// SelfAttention_3032246911287
// MI455X (gfx1250) — matrix operand order repaired and checked by reading, not compiled, not run on hardware
//
#include <hip/hip_runtime.h>
#include <stdint.h>

typedef _Float16 half_t;
typedef __attribute__((ext_vector_type(16))) _Float16 v16h;
typedef __attribute__((ext_vector_type(8)))  _Float16 v8h;
typedef __attribute__((ext_vector_type(8)))  float    v8f;
typedef __attribute__((ext_vector_type(4)))  float    f32x4;
typedef __attribute__((ext_vector_type(4)))  unsigned u32x4;
typedef __attribute__((ext_vector_type(8)))  int      i32x8;
typedef __attribute__((ext_vector_type(4)))  int      i32x4;

#define BD 32      // batch
#define CD 256     // channels
#define ND 1024    // pixels (32*32)
#define C8 32      // q/k head dim
#define K1_WSTRIDE 264   // f16 row stride for LDS tiles in kernel 1 (pad vs 256)

// ---------------- WMMA wrapper (gfx1250, wave32) ----------------
static __device__ __forceinline__ v8f wmma_f16(v16h a, v16h b, v8f c) {
  return __builtin_amdgcn_wmma_f32_16x16x32_f16(false, a, false, b, (short)0, c,
                                                false, false);
}

// Build a 16-half fragment from two 8-half (16B) chunks.
static __device__ __forceinline__ v16h frag2(const half_t* p0, const half_t* p1) {
  v8h a = *(const v8h*)p0;
  v8h b = *(const v8h*)p1;
  v16h r;
#pragma unroll
  for (int i = 0; i < 8; ++i) { r[i] = a[i]; r[i + 8] = b[i]; }
  return r;
}

// ---------------- Tensor Data Mover: 2D f16 tile  global -> LDS -------------
// D# packing per CDNA5 ISA (data_size=1 => 2B elts)
static __device__ __forceinline__ void tdm_load_2d(unsigned lds_off, const void* gptr,
                                                   unsigned tensor_d0, unsigned tensor_d1,
                                                   unsigned stride0,
                                                   unsigned tile_d0, unsigned tile_d1) {
  uint64_t ga = (uint64_t)(uintptr_t)gptr;
  u32x4 g0;
  g0[0] = 1u;                                                // count=1, user mode
  g0[1] = lds_off;                                           // lds_addr
  g0[2] = (unsigned)ga;                                      // global_addr[31:0]
  g0[3] = (unsigned)((ga >> 32) & 0x01FFFFFFu) | (2u << 30); // addr[56:32] | type=2
  i32x8 g1;
  g1[0] = (int)(1u << 16);                                   // data_size=1 (2B)
  g1[1] = (int)((tensor_d0 & 0xFFFFu) << 16);
  g1[2] = (int)(((tensor_d0 >> 16) & 0xFFFFu) | ((tensor_d1 & 0xFFFFu) << 16));
  g1[3] = (int)(((tensor_d1 >> 16) & 0xFFFFu) | ((tile_d0 & 0xFFFFu) << 16));
  g1[4] = (int)(tile_d1 & 0xFFFFu);                          // tile_dim1, tile_dim2=0
  g1[5] = (int)stride0;                                      // tensor_dim0_stride
  g1[6] = 0;
  g1[7] = 0;
  i32x4 z4 = {0, 0, 0, 0};
  i32x8 z8 = {0, 0, 0, 0, 0, 0, 0, 0};
  __builtin_amdgcn_tensor_load_to_lds(g0, g1, z4, z4, z8, 0);
}

// =====================================================================
// Kernel 1: fused QKV projection.  [320,256] x [256, 128-pixel tile]
// =====================================================================
__global__ __launch_bounds__(256) void qkv_proj_k(
    const float* __restrict__ x,
    const float* __restrict__ Wq, const float* __restrict__ bq,
    const float* __restrict__ Wk, const float* __restrict__ bk,
    const float* __restrict__ Wv, const float* __restrict__ bv,
    half_t* __restrict__ Qb, half_t* __restrict__ KbT, half_t* __restrict__ Vb) {
  extern __shared__ char smem[];
  half_t* Wh = (half_t*)smem;                                   // [320][264] f16
  half_t* xs = (half_t*)(smem + 320 * K1_WSTRIDE * 2);          // [128][264] pixel-major
  float*  bs = (float*)(smem + (320 + 128) * K1_WSTRIDE * 2);   // [320] f32

  const int tid  = threadIdx.x;
  const int wave = tid >> 5;
  const int lane = tid & 31;
  const int lh   = lane >> 4;
  const int ll   = lane & 15;
  const int b    = blockIdx.x >> 3;
  const int p0   = (blockIdx.x & 7) * 128;

  for (int idx = tid; idx < 320 * 256; idx += 256) {
    int row = idx >> 8, col = idx & 255;
    float w = (row < 32) ? Wq[row * 256 + col]
            : (row < 64) ? Wk[(row - 32) * 256 + col]
                         : Wv[(row - 64) * 256 + col];
    Wh[row * K1_WSTRIDE + col] = (half_t)w;
  }
  for (int idx = tid; idx < 320; idx += 256)
    bs[idx] = (idx < 32) ? bq[idx] : (idx < 64) ? bk[idx - 32] : bv[idx - 64];
  for (int idx = tid; idx < 256 * 128; idx += 256) {
    int c = idx >> 7, p = idx & 127;
    xs[p * K1_WSTRIDE + c] = (half_t)x[((size_t)b * CD + c) * ND + p0 + p];
  }
  __syncthreads();

  // Per-wave pixel-tile column is fixed (t%8==wave): B fragments loop-invariant.
  const int c0 = wave * 16;
  v16h bfr[8];
#pragma unroll
  for (int k8 = 0; k8 < 8; ++k8) {
    const half_t* bp = xs + (c0 + ll) * K1_WSTRIDE + k8 * 32 + lh * 8;
    bfr[k8] = frag2(bp, bp + 16);
  }

  for (int rt = 0; rt < 20; ++rt) {      // 20 output-channel tiles per wave
    const int r0 = rt * 16;
    // Batch all A-fragment LDS loads, then chain the 8 WMMAs.
    v16h af[8];
#pragma unroll
    for (int k8 = 0; k8 < 8; ++k8) {
      const half_t* ap = Wh + (r0 + ll) * K1_WSTRIDE + k8 * 32 + lh * 8;
      af[k8] = frag2(ap, ap + 16);
    }
    v8f acc = {0.f, 0.f, 0.f, 0.f, 0.f, 0.f, 0.f, 0.f};
#pragma unroll
    for (int k8 = 0; k8 < 8; ++k8) acc = wmma_f16(af[k8], bfr[k8], acc);

    const int n  = p0 + c0 + ll;   // pixel (C/D layout: N = lane%16)
    const int R0 = r0 + 8 * lh;    // first of 8 consecutive channel rows
    f32x4 bl = *(const f32x4*)(bs + R0);
    f32x4 bh = *(const f32x4*)(bs + R0 + 4);
    if (r0 < 32) {                 // Q -> [B][N][32], 8 contiguous halves
      v8h ph;
#pragma unroll
      for (int r = 0; r < 4; ++r) { ph[r] = (half_t)(acc[r] + bl[r]);
                                    ph[r + 4] = (half_t)(acc[r + 4] + bh[r]); }
      *(v8h*)(Qb + ((size_t)b * ND + n) * C8 + R0) = ph;
    } else if (r0 < 64) {          // K^T -> [B][N][32]
      v8h ph;
#pragma unroll
      for (int r = 0; r < 4; ++r) { ph[r] = (half_t)(acc[r] + bl[r]);
                                    ph[r + 4] = (half_t)(acc[r + 4] + bh[r]); }
      *(v8h*)(KbT + ((size_t)b * ND + n) * C8 + (R0 - 32)) = ph;
    } else {                       // V -> [B][256][N] (row stride ND: scalar)
#pragma unroll
      for (int r = 0; r < 8; ++r) {
        float bb = (r < 4) ? bl[r] : bh[r - 4];
        Vb[((size_t)b * CD + (R0 - 64 + r)) * ND + n] = (half_t)(acc[r] + bb);
      }
    }
  }
}

// =====================================================================
// Kernel 2: flash attention + epilogue, double-buffered TDM staging.
// grid = B*8 (128-query blocks), 8 waves; wave: 16 queries x 256 channels.
// =====================================================================
#define KS_OFF(buf) ((unsigned)((buf) * 8192))
#define VS_OFF(buf) ((unsigned)(16384 + (buf) * 65536))
#define PS_OFF      (16384 + 131072)

__global__ __launch_bounds__(256) void attn_k(
    const float* __restrict__ x, const float* __restrict__ gamma,
    const half_t* __restrict__ Qb, const half_t* __restrict__ KbT,
    const half_t* __restrict__ Vb, float* __restrict__ out) {
  extern __shared__ char smem[];

  const int tid  = threadIdx.x;
  const int wave = tid >> 5;
  const int lane = tid & 31;
  const int lh   = lane >> 4;
  const int ll   = lane & 15;
  const int b    = blockIdx.x >> 3;
  const int qblk = blockIdx.x & 7;
  const int q0   = qblk * 128 + wave * 16;

  const unsigned lds_base = (unsigned)(uintptr_t)smem;  // LDS aperture offset
  half_t* Psw = (half_t*)(smem + PS_OFF) + wave * (16 * 128);

  // Q fragment lives in registers for the whole kernel (A: 16 q x 32 d)
  const half_t* qrow = Qb + ((size_t)b * ND + q0 + ll) * C8 + lh * 8;
  const v16h qfrag = frag2(qrow, qrow + 16);

  float mrow[8], lrow[8];
#pragma unroll
  for (int r = 0; r < 8; ++r) { mrow[r] = -3.0e38f; lrow[r] = 0.0f; }
  v8f o[16];
#pragma unroll
  for (int t = 0; t < 16; ++t) o[t] = (v8f){0.f, 0.f, 0.f, 0.f, 0.f, 0.f, 0.f, 0.f};

  // Prologue: async-stage key block 0 into buffer 0 (wave 0 drives the TDM).
  if (tid < 32) {
    tdm_load_2d(lds_base + KS_OFF(0), KbT + ((size_t)b * ND + 0) * C8,
                C8, ND, C8, C8, 128);
    tdm_load_2d(lds_base + VS_OFF(0), Vb + (size_t)b * CD * ND + 0,
                ND, CD, ND, 128, CD);
  }

  for (int jb = 0; jb < 8; ++jb) {     // 8 key blocks of 128 keys
    const int buf = jb & 1;
    if (tid < 32) {
      if (jb < 7) {                    // prefetch next block into other buffer
        tdm_load_2d(lds_base + KS_OFF(buf ^ 1),
                    KbT + ((size_t)b * ND + (jb + 1) * 128) * C8,
                    C8, ND, C8, C8, 128);
        tdm_load_2d(lds_base + VS_OFF(buf ^ 1),
                    Vb + (size_t)b * CD * ND + (jb + 1) * 128,
                    ND, CD, ND, 128, CD);
        __builtin_amdgcn_s_wait_tensorcnt(2);  // in-order: current block done
      } else {
        __builtin_amdgcn_s_wait_tensorcnt(0);
      }
    }
    __syncthreads();                   // current tiles visible; prev reads done

    const half_t* KsT = (const half_t*)(smem + KS_OFF(buf)); // [128 keys][32]
    const half_t* Vs  = (const half_t*)(smem + VS_OFF(buf)); // [256 d][128 keys]

    // S = Q K^T : batch all K-fragment loads, then 8 WMMAs (K=32=d_k).
    v16h kf[8];
#pragma unroll
    for (int j = 0; j < 8; ++j) {
      const half_t* kp = KsT + (j * 16 + ll) * C8 + lh * 8;
      kf[j] = frag2(kp, kp + 16);
    }
    v8f s[8];
#pragma unroll
    for (int j = 0; j < 8; ++j) {
      v8f z = {0.f, 0.f, 0.f, 0.f, 0.f, 0.f, 0.f, 0.f};
      s[j] = wmma_f16(qfrag, kf[j], z);
    }

    // ---- online softmax (rows spread over 16 lanes + 8 j-tiles) ----
    float alpha[8];
#pragma unroll
    for (int r = 0; r < 8; ++r) {
      float v = s[0][r];
#pragma unroll
      for (int j = 1; j < 8; ++j) v = fmaxf(v, s[j][r]);
#pragma unroll
      for (int m = 1; m < 16; m <<= 1) v = fmaxf(v, __shfl_xor(v, m, 32));
      float mnew = fmaxf(mrow[r], v);
      alpha[r] = __expf(mrow[r] - mnew);
      mrow[r] = mnew;
    }
#pragma unroll
    for (int j = 0; j < 8; ++j)
#pragma unroll
      for (int r = 0; r < 8; ++r) s[j][r] = __expf(s[j][r] - mrow[r]);
#pragma unroll
    for (int r = 0; r < 8; ++r) {
      float v = 0.f;
#pragma unroll
      for (int j = 0; j < 8; ++j) v += s[j][r];
#pragma unroll
      for (int m = 1; m < 16; m <<= 1) v += __shfl_xor(v, m, 32);
      lrow[r] = lrow[r] * alpha[r] + v;
    }
#pragma unroll
    for (int t = 0; t < 16; ++t)
#pragma unroll
      for (int r = 0; r < 8; ++r) o[t][r] *= alpha[r];

    // Re-layout P (C layout) -> per-wave LDS -> A fragments (no barrier needed)
#pragma unroll
    for (int j = 0; j < 8; ++j)
#pragma unroll
      for (int r = 0; r < 8; ++r)
        Psw[(r + 8 * lh) * 128 + j * 16 + ll] = (half_t)s[j][r];

    // P fragments depend only on kk: hoist out of the 16 d-tile loop.
    v16h pf[4];
#pragma unroll
    for (int k4 = 0; k4 < 4; ++k4) {
      const half_t* pp = Psw + ll * 128 + k4 * 32 + lh * 8;
      pf[k4] = frag2(pp, pp + 16);
    }

    // O += P * V^T  (16 d-tiles x 4 WMMA over the 128-key block)
#pragma unroll
    for (int dt = 0; dt < 16; ++dt) {
      v16h vf[4];
#pragma unroll
      for (int k4 = 0; k4 < 4; ++k4) {
        const half_t* vp = Vs + (dt * 16 + ll) * 128 + k4 * 32 + lh * 8;
        vf[k4] = frag2(vp, vp + 16);
      }
      v8f c = o[dt];
#pragma unroll
      for (int k4 = 0; k4 < 4; ++k4) c = wmma_f16(pf[k4], vf[k4], c);
      o[dt] = c;
    }
    __syncthreads();                   // done reading buf before TDM overwrites
  }

  // Epilogue: out = gamma * (O / l) + x   (8 consecutive q per lane -> f32x4)
  const float g = gamma[0];
  float inv[8];
#pragma unroll
  for (int r = 0; r < 8; ++r) inv[r] = 1.0f / lrow[r];
#pragma unroll
  for (int dt = 0; dt < 16; ++dt) {
    const int d = dt * 16 + ll;
    const size_t base = ((size_t)b * CD + d) * ND + (size_t)(q0 + 8 * lh);
    f32x4 x0 = *(const f32x4*)(x + base);
    f32x4 x1 = *(const f32x4*)(x + base + 4);
    f32x4 y0, y1;
#pragma unroll
    for (int r = 0; r < 4; ++r) {
      y0[r] = g * (o[dt][r] * inv[r]) + x0[r];
      y1[r] = g * (o[dt][r + 4] * inv[r + 4]) + x1[r];
    }
    *(f32x4*)(out + base) = y0;
    *(f32x4*)(out + base + 4) = y1;
  }
}

// =====================================================================
extern "C" void kernel_launch(void* const* d_in, const int* in_sizes, int n_in,
                              void* d_out, int out_size, void* d_ws, size_t ws_size,
                              hipStream_t stream) {
  const float* x  = (const float*)d_in[0];
  const float* Wq = (const float*)d_in[1];
  const float* bq = (const float*)d_in[2];
  const float* Wk = (const float*)d_in[3];
  const float* bk = (const float*)d_in[4];
  const float* Wv = (const float*)d_in[5];
  const float* bv = (const float*)d_in[6];
  const float* gm = (const float*)d_in[7];
  float* out = (float*)d_out;

  // Workspace: Q [B][1024][32] f16 | K^T [B][1024][32] f16 | V [B][256][1024] f16
  half_t* Qb  = (half_t*)d_ws;
  half_t* KbT = (half_t*)((char*)d_ws + (size_t)2 * 1024 * 1024);
  half_t* Vb  = (half_t*)((char*)d_ws + (size_t)4 * 1024 * 1024);

  const size_t smem1 = (size_t)(320 + 128) * K1_WSTRIDE * 2 + 320 * 4;  // ~233KB
  qkv_proj_k<<<dim3(BD * 8), dim3(256), smem1, stream>>>(x, Wq, bq, Wk, bk, Wv, bv,
                                                         Qb, KbT, Vb);
  const size_t smem2 = PS_OFF + 8 * 16 * 128 * 2;                       // 176KB
  attn_k<<<dim3(BD * 8), dim3(256), smem2, stream>>>(x, gm, Qb, KbT, Vb, out);
}
